// ORViT_81982335746430
// MI455X (gfx1250) — compile-verified, matrix operand order checked by reading, not run on hardware
//
#include <hip/hip_runtime.h>
#include <hip/hip_bf16.h>

typedef __attribute__((ext_vector_type(16))) _Float16 v16h;
typedef __attribute__((ext_vector_type(8)))  float    v8f;

#define DEV __device__ __forceinline__

// Problem constants (fixed by setup_inputs)
#define BSZ   2
#define TT    8
#define HH    14
#define WW    14
#define DD    768
#define OO    6
#define PP    202            // tokens per frame (196 patch + 6 obj)
#define NN    1616           // TT*PP
#define NTOK  3232           // BSZ*NN
#define NHEAD 12
#define DH    64
#define ROWS1 1536           // BSZ*TT*OO*16 roi sample rows
#define SCALE 0.125f         // dh^-0.5

struct FragH {
  union { v16h v; _Float16 h[16]; unsigned u[8]; };
};

// K-index map for 16-bit A/B fragments (ISA 7.12.2):
// half-wave hs owns K-octet {hs*8..hs*8+7, 16+hs*8..}; VGPR j packs pair (k,k+1).
DEV int kmap(int j, int hs) { return (j < 4) ? (hs * 8 + 2 * j) : (16 + hs * 8 + 2 * (j - 4)); }

// Load fragment whose per-lane row pointer has been precomputed:
// p = &Mat[row_of_this_lane][k + hs*8]; pairs (k,k+1) contiguous -> b128-coalescible.
DEV void ldfrag(FragH& fr, const _Float16* p) {
#pragma unroll
  for (int j = 0; j < 4; ++j) {
    fr.u[j]     = *(const unsigned*)(p + 2 * j);
    fr.u[4 + j] = *(const unsigned*)(p + 16 + 2 * j);
  }
}

// Row-major fragment (base = &Mat[row0][k0], ld = row stride in elements).
DEV void load_rowpair_frag(FragH& fr, const _Float16* base, int ld) {
  int lane = threadIdx.x & 31;
  int r = lane & 15, hs = lane >> 4;
  ldfrag(fr, base + (size_t)r * ld + hs * 8);
}

DEV v8f wmma_f16(const FragH& a, const FragH& b, v8f c) {
  return __builtin_amdgcn_wmma_f32_16x16x32_f16(false, a.v, false, b.v, (short)0, c, false, false);
}

// ---------------------------------------------------------------------------
// Weight convert + transpose: src f32 [K][N] -> dst f16 [N][K]
// ---------------------------------------------------------------------------
__global__ void k_wt(const float* __restrict__ src, _Float16* __restrict__ dst, int K, int N) {
  int i = blockIdx.x * 256 + threadIdx.x;
  if (i >= K * N) return;
  int kk = i / N, n = i % N;
  dst[(size_t)n * K + kk] = (_Float16)src[i];
}

// ---------------------------------------------------------------------------
// ROI bilinear sampling: one block per sample row (b,t,o,sy,sx) -> S1 f16 [1536][768]
// ---------------------------------------------------------------------------
__global__ void k_roi(const float* __restrict__ x, const float* __restrict__ meta,
                      _Float16* __restrict__ S1) {
  int row = blockIdx.x;                        // ((b*T+t)*O+o)*16 + s
  int s = row & 15, o = (row >> 4) % OO, t = (row / (16 * OO)) % TT, b = row / (16 * OO * TT);
  int sy = s >> 2, sx = s & 3;
  const float* bx = meta + (((size_t)b * OO + o) * TT + t) * 4;   // metadata (B,O,T,4)
  const float sc = (float)WW / 224.0f;
  float x1 = bx[0] * sc, y1 = bx[1] * sc, x2 = bx[2] * sc, y2 = bx[3] * sc;
  float xs = x1 + ((sx + 0.5f) / 4.0f) * (x2 - x1) - 0.5f;
  float ys = y1 + ((sy + 0.5f) / 4.0f) * (y2 - y1) - 0.5f;
  float x0f = floorf(xs), y0f = floorf(ys);
  float lx = xs - x0f, ly = ys - y0f;
  int x0 = (int)x0f, y0 = (int)y0f;
  int x1i = x0 + 1, y1i = y0 + 1;
  float m00 = (y0 >= 0 && y0 < HH && x0 >= 0 && x0 < WW) ? 1.f : 0.f;
  float m01 = (y0 >= 0 && y0 < HH && x1i >= 0 && x1i < WW) ? 1.f : 0.f;
  float m10 = (y1i >= 0 && y1i < HH && x0 >= 0 && x0 < WW) ? 1.f : 0.f;
  float m11 = (y1i >= 0 && y1i < HH && x1i >= 0 && x1i < WW) ? 1.f : 0.f;
  int yc0 = min(max(y0, 0), HH - 1), yc1 = min(max(y1i, 0), HH - 1);
  int xc0 = min(max(x0, 0), WW - 1), xc1 = min(max(x1i, 0), WW - 1);
  const float* base = x + ((size_t)b * (1 + TT * HH * WW) + 1 + t * HH * WW) * DD;
  float w00 = (1 - ly) * (1 - lx), w01 = (1 - ly) * lx, w10 = ly * (1 - lx), w11 = ly * lx;
  for (int c = threadIdx.x; c < DD; c += 256) {
    float v00 = base[(yc0 * WW + xc0) * DD + c] * m00;
    float v01 = base[(yc0 * WW + xc1) * DD + c] * m01;
    float v10 = base[(yc1 * WW + xc0) * DD + c] * m10;
    float v11 = base[(yc1 * WW + xc1) * DD + c] * m11;
    S1[(size_t)row * DD + c] = (_Float16)(v00 * w00 + v01 * w01 + v10 * w10 + v11 * w11);
  }
}

// ---------------------------------------------------------------------------
// WMMA GEMM: C[M][N] = act(scale * A[M][K] @ Bt[N][K]^T)
// block = 256 threads (8 waves, 2x4), each wave computes a 32x32 macro-tile
// (2x2 WMMA tiles) -> 64x128 supertile. Software-pipelined ping-pong buffers
// with COMPILE-TIME buffer selection (K unrolled by 64) so all fragments stay
// in VGPRs (no scratch/LDS spills). Requires N%128==0, K%64==0.
// ---------------------------------------------------------------------------
__global__ void k_gemm(const _Float16* __restrict__ A, const _Float16* __restrict__ Bt,
                       _Float16* __restrict__ outH, float* __restrict__ outF,
                       int M, int N, int K, float scale, int relu) {
  int wave = threadIdx.x >> 5;
  int m0 = blockIdx.y * 64 + (wave >> 2) * 32;
  int n0 = blockIdx.x * 128 + (wave & 3) * 32;
  int maxrow = M - 1;
  int lane = threadIdx.x & 31;
  int r = lane & 15, hs = lane >> 4;

  // Per-lane row pointers, hoisted out of the K loop. A rows clamped so
  // addresses stay valid on ragged M edges (EXEC must be all-ones for WMMA).
  const _Float16* arow[2];
  const _Float16* brow[2];
#pragma unroll
  for (int t = 0; t < 2; ++t) {
    int row = m0 + t * 16 + r; if (row > maxrow) row = maxrow;
    arow[t] = A + (size_t)row * K + hs * 8;
    brow[t] = Bt + (size_t)(n0 + t * 16 + r) * K + hs * 8;
  }

  FragH a0[2], b0[2], a1[2], b1[2];
  v8f acc[2][2] = {{{}, {}}, {{}, {}}};

#pragma unroll
  for (int t = 0; t < 2; ++t) { ldfrag(a0[t], arow[t]); ldfrag(b0[t], brow[t]); }

  for (int k = 0; k < K; k += 64) {
    // stage buffer 1 (k+32 always < K since K % 64 == 0)
#pragma unroll
    for (int t = 0; t < 2; ++t) { ldfrag(a1[t], arow[t] + k + 32); ldfrag(b1[t], brow[t] + k + 32); }
    if (k + 64 < K) {
      __builtin_prefetch(arow[0] + k + 128, 0, 1);
      __builtin_prefetch(brow[0] + k + 128, 0, 1);
    }
#pragma unroll
    for (int mt = 0; mt < 2; ++mt)
#pragma unroll
      for (int nt = 0; nt < 2; ++nt)
        acc[mt][nt] = wmma_f16(a0[mt], b0[nt], acc[mt][nt]);
    // stage buffer 0 for next iteration while buffer-1 WMMAs issue
    if (k + 64 < K) {
#pragma unroll
      for (int t = 0; t < 2; ++t) { ldfrag(a0[t], arow[t] + k + 64); ldfrag(b0[t], brow[t] + k + 64); }
    }
#pragma unroll
    for (int mt = 0; mt < 2; ++mt)
#pragma unroll
      for (int nt = 0; nt < 2; ++nt)
        acc[mt][nt] = wmma_f16(a1[mt], b1[nt], acc[mt][nt]);
  }

  int n = lane & 15;
#pragma unroll
  for (int mt = 0; mt < 2; ++mt)
#pragma unroll
    for (int nt = 0; nt < 2; ++nt)
#pragma unroll
      for (int j = 0; j < 8; ++j) {
        int row = m0 + mt * 16 + j + hs * 8;
        if (row >= M) continue;
        float v = acc[mt][nt][j] * scale;
        if (relu) v = fmaxf(v, 0.f);
        size_t idx = (size_t)row * N + n0 + nt * 16 + n;
        if (outH) outH[idx] = (_Float16)v; else outF[idx] = v;
      }
}

// ---------------------------------------------------------------------------
// Max-pool MLP output over 16 spatial samples -> obj f32 (B,T,O,768)
// ---------------------------------------------------------------------------
__global__ void k_maxpool(const _Float16* __restrict__ M2, float* __restrict__ obj) {
  int g = blockIdx.x;  // b*T*O group
  const _Float16* base = M2 + (size_t)g * 16 * DD;
  for (int c = threadIdx.x; c < DD; c += 256) {
    float m = (float)base[c];
#pragma unroll
    for (int s = 1; s < 16; ++s) m = fmaxf(m, (float)base[(size_t)s * DD + c]);
    obj[(size_t)g * DD + c] = m;
  }
}

// ---------------------------------------------------------------------------
// Assemble allt row + LayerNorm -> f16 [NTOK][768]
// ---------------------------------------------------------------------------
__global__ void k_ln(const float* __restrict__ x, const float* __restrict__ obj,
                     const float* __restrict__ g, const float* __restrict__ be,
                     _Float16* __restrict__ out) {
  __shared__ float buf[DD];
  __shared__ float red[256];
  int row = blockIdx.x;                 // b*NN + t*PP + j
  int b = row / NN, r = row % NN, t = r / PP, j = r % PP;
  const float* src = (j < HH * WW)
      ? x + ((size_t)b * (1 + TT * HH * WW) + 1 + t * HH * WW + j) * DD
      : obj + (((size_t)(b * TT + t) * OO) + (j - HH * WW)) * DD;
  int tid = threadIdx.x;
  float s = 0.f;
  for (int i = tid; i < DD; i += 256) { float v = src[i]; buf[i] = v; s += v; }
  red[tid] = s; __syncthreads();
  for (int st = 128; st > 0; st >>= 1) { if (tid < st) red[tid] += red[tid + st]; __syncthreads(); }
  float mu = red[0] / (float)DD; __syncthreads();
  float vs = 0.f;
  for (int i = tid; i < DD; i += 256) { float dv = buf[i] - mu; vs += dv * dv; }
  red[tid] = vs; __syncthreads();
  for (int st = 128; st > 0; st >>= 1) { if (tid < st) red[tid] += red[tid + st]; __syncthreads(); }
  float rstd = rsqrtf(red[0] / (float)DD + 1e-5f);
  for (int i = tid; i < DD; i += 256)
    out[(size_t)row * DD + i] = (_Float16)((buf[i] - mu) * rstd * g[i] + be[i]);
}

// ---------------------------------------------------------------------------
// Stage-1 trajectory attention: per (b,h,f, 16-row q tile):
//   logits = q@k_f^T (WMMA) -> LDS softmax over P=202 -> traj_f = probs@v_f (WMMA)
// traj layout: (B,N,F,C) f16.  block = 1 wave (32 threads).
// ---------------------------------------------------------------------------
__global__ void k_attn1(const _Float16* __restrict__ q, const _Float16* __restrict__ kmat,
                        const _Float16* __restrict__ vmat, _Float16* __restrict__ traj) {
  __shared__ float    lg[16][224];
  __shared__ _Float16 pr[16][224];
  int mt = blockIdx.x;
  int idx = blockIdx.y;
  int f = idx & 7, h = (idx >> 3) % NHEAD, b = idx / (NHEAD * TT);
  int lane = threadIdx.x;
  int nloc = lane & 15, hs = lane >> 4;
  const _Float16* qb = q + ((size_t)b * NN + mt * 16) * DD + h * DH;   // q pre-scaled

  // ---- logits (14 chunks of 16 keys; K-loop over dh=64 in 2 steps) ----
  for (int pc = 0; pc < 14; ++pc) {
    v8f c = {};
    int key = pc * 16 + nloc; if (key > PP - 1) key = PP - 1;
    const _Float16* kp = kmat + ((size_t)b * NN + f * PP + key) * DD + h * DH;
#pragma unroll
    for (int kk = 0; kk < DH; kk += 32) {
      FragH a, bf;
      load_rowpair_frag(a, qb + kk, DD);
      ldfrag(bf, kp + kk + hs * 8);
      c = wmma_f16(a, bf, c);
    }
#pragma unroll
    for (int j = 0; j < 8; ++j) lg[j + hs * 8][pc * 16 + nloc] = c[j];
  }
  __syncthreads();

  // ---- softmax over p<202; zero-pad to 224 ----
  if (lane < 16) {
    float m = -1e30f;
    for (int p = 0; p < PP; ++p) m = fmaxf(m, lg[lane][p]);
    float s = 0.f;
    for (int p = 0; p < PP; ++p) { float e = __expf(lg[lane][p] - m); s += e; lg[lane][p] = e; }
    float inv = 1.f / s;
    for (int p = 0; p < PP; ++p) pr[lane][p] = (_Float16)(lg[lane][p] * inv);
    for (int p = PP; p < 224; ++p) pr[lane][p] = (_Float16)0.f;
  }
  __syncthreads();

  // ---- traj tile = probs(16x224) @ v_f(224x64), padded probs are zero ----
  v8f acc[4] = {{}, {}, {}, {}};
  for (int kc = 0; kc < 7; ++kc) {
    FragH a;
    load_rowpair_frag(a, &pr[0][kc * 32], 224);
#pragma unroll
    for (int nt = 0; nt < 4; ++nt) {
      FragH bb;
      int col = h * DH + nt * 16 + nloc;
#pragma unroll
      for (int j = 0; j < 8; ++j) {
        int k0 = kmap(j, hs);
        int p0 = kc * 32 + k0;     if (p0 > PP - 1) p0 = PP - 1;
        int p1 = kc * 32 + k0 + 1; if (p1 > PP - 1) p1 = PP - 1;
        bb.h[2 * j]     = vmat[((size_t)b * NN + f * PP + p0) * DD + col];
        bb.h[2 * j + 1] = vmat[((size_t)b * NN + f * PP + p1) * DD + col];
      }
      acc[nt] = wmma_f16(a, bb, acc[nt]);
    }
  }
#pragma unroll
  for (int nt = 0; nt < 4; ++nt)
#pragma unroll
    for (int j = 0; j < 8; ++j) {
      int tok = mt * 16 + j + hs * 8;
      traj[(((size_t)b * NN + tok) * TT + f) * DD + h * DH + nt * 16 + nloc] = (_Float16)acc[nt][j];
    }
}

// ---------------------------------------------------------------------------
// x_diag[b,n,c] = traj[b,n, n/P, c]
// ---------------------------------------------------------------------------
__global__ void k_diag(const _Float16* __restrict__ traj, _Float16* __restrict__ xdiag) {
  size_t i = (size_t)blockIdx.x * 256 + threadIdx.x;
  if (i >= (size_t)NTOK * DD) return;
  size_t bn = i / DD; int c = (int)(i % DD);
  int n = (int)(bn % NN);
  int f = n / PP;
  xdiag[i] = traj[(bn * TT + f) * DD + c];
}

// ---------------------------------------------------------------------------
// Fused stage-2: per token n: K2V2(8x1536) = traj[b,n](8x768) @ w_kv (WMMA, in LDS),
// then a2 = softmax_f(q2 . k2), o = sum_f a2*v2. Avoids 80MB k2v2 materialization.
// block = 256 threads (8 waves), grid = NTOK.
// ---------------------------------------------------------------------------
__global__ void k_attn2(const _Float16* __restrict__ traj, const _Float16* __restrict__ q2,
                        const _Float16* __restrict__ wT_kv, _Float16* __restrict__ o) {
  __shared__ _Float16 ta[TT][DD];        // 12 KB
  __shared__ _Float16 kv[TT][2 * DD];    // 24 KB  (cols 0..767 = k2, 768..1535 = v2)
  __shared__ float    lg[NHEAD * TT];
  __shared__ float    a2[NHEAD][TT];
  int bn = blockIdx.x;
  const _Float16* tg = traj + (size_t)bn * TT * DD;
  for (int i = threadIdx.x; i < TT * DD / 2; i += 256)
    ((unsigned*)&ta[0][0])[i] = ((const unsigned*)tg)[i];
  __syncthreads();

  int wave = threadIdx.x >> 5, lane = threadIdx.x & 31;
  int nloc = lane & 15, hs = lane >> 4, r = lane & 15;
  for (int nt = wave; nt < (2 * DD) / 16; nt += 8) {
    v8f c = {};
    for (int kc = 0; kc < DD / 32; ++kc) {
      FragH a, b;
      if (r < TT) {
        ldfrag(a, &ta[r][kc * 32 + hs * 8]);
      } else {
#pragma unroll
        for (int j = 0; j < 8; ++j) a.u[j] = 0u;
      }
      load_rowpair_frag(b, wT_kv + (size_t)(nt * 16) * DD + kc * 32, DD);
      c = wmma_f16(a, b, c);
    }
    if (hs == 0) {  // rows m=j (0..7) hold f=0..7
#pragma unroll
      for (int j = 0; j < 8; ++j) kv[j][nt * 16 + nloc] = (_Float16)c[j];
    }
  }
  __syncthreads();

  const _Float16* q2r = q2 + (size_t)bn * DD;   // q2 pre-scaled
  if (threadIdx.x < NHEAD * TT) {
    int h = threadIdx.x >> 3, f = threadIdx.x & 7;
    float s = 0.f;
    for (int dI = 0; dI < DH; ++dI) s += (float)q2r[h * DH + dI] * (float)kv[f][h * DH + dI];
    lg[threadIdx.x] = s;
  }
  __syncthreads();
  if (threadIdx.x < NHEAD) {
    int h = threadIdx.x;
    float m = -1e30f;
    for (int f = 0; f < TT; ++f) m = fmaxf(m, lg[h * TT + f]);
    float s = 0.f;
    for (int f = 0; f < TT; ++f) { float e = __expf(lg[h * TT + f] - m); a2[h][f] = e; s += e; }
    float inv = 1.f / s;
    for (int f = 0; f < TT; ++f) a2[h][f] *= inv;
  }
  __syncthreads();
  for (int c = threadIdx.x; c < DD; c += 256) {
    int h = c >> 6;
    float s = 0.f;
#pragma unroll
    for (int f = 0; f < TT; ++f) s += a2[h][f] * (float)kv[f][DD + c];
    o[(size_t)bn * DD + c] = (_Float16)s;
  }
}

// ---------------------------------------------------------------------------
// Residual scatter: out = x + concat([cls, proj(patch)+b_proj]); obj tokens dropped.
// ---------------------------------------------------------------------------
__global__ void k_residual(const float* __restrict__ x, const float* __restrict__ projf,
                           const float* __restrict__ bproj, float* __restrict__ out) {
  size_t i = (size_t)blockIdx.x * 256 + threadIdx.x;
  const size_t total = (size_t)BSZ * (1 + TT * HH * WW) * DD;
  if (i >= total) return;
  int c = (int)(i % DD);
  size_t tok = i / DD;
  int b = (int)(tok / (1 + TT * HH * WW));
  int t1 = (int)(tok % (1 + TT * HH * WW));
  float xv = x[i];
  if (t1 == 0) { out[i] = xv + xv; return; }     // cls: x + cls_tok
  int pi = t1 - 1, t = pi / (HH * WW), hw = pi % (HH * WW);
  size_t row = (size_t)b * NN + t * PP + hw;
  out[i] = xv + projf[row * DD + c] + bproj[c];
}

// ---------------------------------------------------------------------------
extern "C" void kernel_launch(void* const* d_in, const int* in_sizes, int n_in,
                              void* d_out, int out_size, void* d_ws, size_t ws_size,
                              hipStream_t stream) {
  const float* x      = (const float*)d_in[0];
  const float* meta   = (const float*)d_in[1];
  const float* p2d_w1 = (const float*)d_in[2];
  const float* p2d_w2 = (const float*)d_in[3];
  const float* ln1_g  = (const float*)d_in[4];
  const float* ln1_b  = (const float*)d_in[5];
  const float* w_qkv  = (const float*)d_in[6];
  const float* w_q    = (const float*)d_in[7];
  const float* w_kv   = (const float*)d_in[8];
  const float* w_proj = (const float*)d_in[9];
  const float* b_proj = (const float*)d_in[10];
  float* out = (float*)d_out;

  char* ws = (char*)d_ws;
  size_t off = 0;
  auto alloc = [&](size_t bytes) { size_t o = off; off = (off + bytes + 255) & ~(size_t)255; return o; };

  _Float16* wT_qkv  = (_Float16*)(ws + alloc((size_t)3 * DD * DD * 2));
  _Float16* wT_q    = (_Float16*)(ws + alloc((size_t)DD * DD * 2));
  _Float16* wT_kv   = (_Float16*)(ws + alloc((size_t)2 * DD * DD * 2));
  _Float16* wT_proj = (_Float16*)(ws + alloc((size_t)DD * DD * 2));
  _Float16* wT_p1   = (_Float16*)(ws + alloc((size_t)(DD / 2) * DD * 2));
  _Float16* wT_p2   = (_Float16*)(ws + alloc((size_t)DD * (DD / 2) * 2));
  _Float16* S1      = (_Float16*)(ws + alloc((size_t)ROWS1 * DD * 2));
  _Float16* Hbuf    = (_Float16*)(ws + alloc((size_t)ROWS1 * (DD / 2) * 2));
  _Float16* M2      = (_Float16*)(ws + alloc((size_t)ROWS1 * DD * 2));
  float*    obj     = (float*)   (ws + alloc((size_t)BSZ * TT * OO * DD * 4));
  _Float16* Aln     = (_Float16*)(ws + alloc((size_t)NTOK * DD * 2));
  _Float16* qbuf    = (_Float16*)(ws + alloc((size_t)NTOK * DD * 2));
  _Float16* kbuf    = (_Float16*)(ws + alloc((size_t)NTOK * DD * 2));
  _Float16* vbuf    = (_Float16*)(ws + alloc((size_t)NTOK * DD * 2));
  _Float16* traj    = (_Float16*)(ws + alloc((size_t)NTOK * TT * DD * 2));
  _Float16* xdiag   = (_Float16*)(ws + alloc((size_t)NTOK * DD * 2));
  _Float16* q2buf   = (_Float16*)(ws + alloc((size_t)NTOK * DD * 2));
  _Float16* obuf    = (_Float16*)(ws + alloc((size_t)NTOK * DD * 2));
  float*    projf   = (float*)   (ws + alloc((size_t)NTOK * DD * 4));

  dim3 blk(256);
  auto g1 = [](int n) { return dim3((unsigned)((n + 255) / 256)); };
  auto gg = [](int M, int N) { return dim3((unsigned)(N / 128), (unsigned)((M + 63) / 64)); };

  // Weight transpose/convert to f16 [N][K]
  k_wt<<<g1(DD * 3 * DD), blk, 0, stream>>>(w_qkv,  wT_qkv,  DD, 3 * DD);
  k_wt<<<g1(DD * DD),     blk, 0, stream>>>(w_q,    wT_q,    DD, DD);
  k_wt<<<g1(DD * 2 * DD), blk, 0, stream>>>(w_kv,   wT_kv,   DD, 2 * DD);
  k_wt<<<g1(DD * DD),     blk, 0, stream>>>(w_proj, wT_proj, DD, DD);
  k_wt<<<g1(DD * DD / 2), blk, 0, stream>>>(p2d_w1, wT_p1,   DD, DD / 2);
  k_wt<<<g1(DD * DD / 2), blk, 0, stream>>>(p2d_w2, wT_p2,   DD / 2, DD);

  // ROI sampling + MLP + maxpool
  k_roi<<<dim3(ROWS1), blk, 0, stream>>>(x, meta, S1);
  k_gemm<<<gg(ROWS1, DD / 2), blk, 0, stream>>>(S1, wT_p1, Hbuf, nullptr, ROWS1, DD / 2, DD, 1.f, 1);
  k_gemm<<<gg(ROWS1, DD), blk, 0, stream>>>(Hbuf, wT_p2, M2, nullptr, ROWS1, DD, DD / 2, 1.f, 1);
  k_maxpool<<<dim3(BSZ * TT * OO), blk, 0, stream>>>(M2, obj);

  // Token assembly + LayerNorm -> f16
  k_ln<<<dim3(NTOK), blk, 0, stream>>>(x, obj, ln1_g, ln1_b, Aln);

  // q/k/v projections (q pre-scaled by dh^-0.5)
  k_gemm<<<gg(NTOK, DD), blk, 0, stream>>>(Aln, wT_qkv,                       qbuf, nullptr, NTOK, DD, DD, SCALE, 0);
  k_gemm<<<gg(NTOK, DD), blk, 0, stream>>>(Aln, wT_qkv + (size_t)DD * DD,     kbuf, nullptr, NTOK, DD, DD, 1.f,   0);
  k_gemm<<<gg(NTOK, DD), blk, 0, stream>>>(Aln, wT_qkv + (size_t)2 * DD * DD, vbuf, nullptr, NTOK, DD, DD, 1.f,   0);

  // Stage-1 attention -> traj (B,N,F,C)
  k_attn1<<<dim3(NN / 16, BSZ * NHEAD * TT), dim3(32), 0, stream>>>(qbuf, kbuf, vbuf, traj);

  // Diagonal gather + q2 projection (pre-scaled)
  k_diag<<<g1(NTOK * DD), blk, 0, stream>>>(traj, xdiag);
  k_gemm<<<gg(NTOK, DD), blk, 0, stream>>>(xdiag, wT_q, q2buf, nullptr, NTOK, DD, DD, SCALE, 0);

  // Fused traj@w_kv + stage-2 attention
  k_attn2<<<dim3(NTOK), blk, 0, stream>>>(traj, q2buf, wT_kv, obuf);

  // Output projection (f32) + residual scatter
  k_gemm<<<gg(NTOK, DD), blk, 0, stream>>>(obuf, wT_proj, nullptr, projf, NTOK, DD, DD, 1.f, 0);
  k_residual<<<g1(BSZ * (1 + TT * HH * WW) * DD), blk, 0, stream>>>(x, projf, b_proj, out);
}
